// ProxyOutlierDetector_24927990186090
// MI455X (gfx1250) — matrix operand order repaired and checked by reading, not compiled, not run on hardware
//
#include <hip/hip_runtime.h>
#include <hip/hip_bf16.h>
#include <math.h>

typedef __attribute__((ext_vector_type(16))) _Float16 v16h;
typedef __attribute__((ext_vector_type(8)))  _Float16 v8h;
typedef __attribute__((ext_vector_type(2)))  _Float16 v2h;
typedef __attribute__((ext_vector_type(8)))  float    v8f;

#define D_DIM   384
#define P_DIM   64
#define KSTEPS  (D_DIM / 32)     // 12
#define NTILES  (P_DIM / 16)     // 4
#define MPIX    128              // pixels per block
#define PITCH   392              // padded row pitch in halves (384 + 8)

// ---------------------------------------------------------------------------
// Kernel 1: normalize proxies and emit B-matrix fragments in the contiguous-K VGPR
// striping this code uses for V_WMMA_F32_16X16X32_F16 (A matches):
//   fragment (nt, kk): lane l in 0..15  -> column n = nt*16+l, K = kk*32+0..15
//                      lane l in 16..31 -> column n = nt*16+(l-16), K = kk*32+16..31
// Each (fragment, lane) owns 16 contiguous halves (32 bytes) in d_ws.
// Total: 4*12*32*16 halves = 48 KB.
// ---------------------------------------------------------------------------
__global__ void proxy_prep_kernel(const float* __restrict__ prox,
                                  _Float16* __restrict__ bfrag) {
    int n = threadIdx.x;
    if (n >= P_DIM) return;
    float sumsq = 0.f;
    #pragma unroll 8
    for (int c = 0; c < D_DIM; ++c) {
        float v = prox[n * D_DIM + c];
        sumsq += v * v;
    }
    float inv = 1.0f / fmaxf(sqrtf(sumsq), 1e-12f);

    int nt = n >> 4;
    int ll = n & 15;
    for (int kk = 0; kk < KSTEPS; ++kk) {
        int baseLo = ((nt * KSTEPS + kk) * 32 + ll) * 16;        // lane ll
        int baseHi = ((nt * KSTEPS + kk) * 32 + 16 + ll) * 16;   // lane 16+ll
        #pragma unroll
        for (int h = 0; h < 16; ++h) {
            bfrag[baseLo + h] = (_Float16)(prox[n * D_DIM + kk * 32 + h] * inv);
            bfrag[baseHi + h] = (_Float16)(prox[n * D_DIM + kk * 32 + 16 + h] * inv);
        }
    }
}

// ---------------------------------------------------------------------------
// Kernel 2: main fused kernel. One block = 128 pixels (contiguous in HW of one
// image), 128 threads = 4 waves.
//  Phase 1: stream x channel-major (coalesced), per-pixel sum of squares in
//           f32, pack raw f16 pairs into LDS [pixel][channel], pitch 392.
//  Phase 2: WMMA f16 GEMM: each wave does 2 M-tiles x 4 N-tiles, K loop of 12.
//  Phase 3: d2 = 2 - 2*acc*inv_norm, min over 64 proxies, sqrt, store.
// ---------------------------------------------------------------------------
__global__ void __launch_bounds__(128)
proxy_min_dist_kernel(const float* __restrict__ x,
                      const _Float16* __restrict__ bfrag,
                      float* __restrict__ out) {
    __shared__ __attribute__((aligned(16))) _Float16 sA[MPIX * PITCH]; // 100352 B
    __shared__ float sInv[MPIX];
    __shared__ float sRed[MPIX * 16];

    const int t   = threadIdx.x;
    const int blk = blockIdx.x;

    // ---- Phase 1: stream + stage -----------------------------------------
    {
        const long  pg  = (long)blk * MPIX + t;       // global pixel id
        const long  b   = pg >> 14;                   // / 16384
        const long  hw  = pg & 16383;
        const float* px = x + b * (long)D_DIM * 16384 + hw;

        float sumsq = 0.f;
        const int row = t * PITCH;
        #pragma unroll 4
        for (int c = 0; c < D_DIM; c += 2) {
            float v0 = px[(long)c * 16384];
            float v1 = px[(long)(c + 1) * 16384];
            sumsq += v0 * v0 + v1 * v1;
            v2h pk;
            pk.x = (_Float16)v0;
            pk.y = (_Float16)v1;
            *reinterpret_cast<v2h*>(&sA[row + c]) = pk;
        }
        sInv[t] = 1.0f / fmaxf(sqrtf(sumsq), 1e-12f);
    }
    __syncthreads();

    // ---- Phase 2: WMMA GEMM ----------------------------------------------
    const int wave = t >> 5;        // 0..3
    const int lane = t & 31;
    const int l15  = lane & 15;
    const bool hi  = lane >= 16;

    // A fragment rows (pixel index within block) for this wave's 2 M-tiles
    const int rowA0 = ((wave * 2 + 0) * 16 + l15) * PITCH;
    const int rowA1 = ((wave * 2 + 1) * 16 + l15) * PITCH;
    const int kadj  = hi ? 16 : 0;  // lanes 16-31 take K 16..31 (contiguous)

    v8f acc[2][NTILES];
    #pragma unroll
    for (int m = 0; m < 2; ++m)
        #pragma unroll
        for (int n = 0; n < NTILES; ++n)
            acc[m][n] = (v8f)(0.0f);

    for (int kk = 0; kk < KSTEPS; ++kk) {
        const int ko = kk * 32 + kadj;
        // A fragments: two 16-byte LDS loads each, 16 contiguous K per lane, same order as B
        v8h a0lo = *reinterpret_cast<const v8h*>(&sA[rowA0 + ko]);
        v8h a0hi = *reinterpret_cast<const v8h*>(&sA[rowA0 + ko + 8]);
        v8h a1lo = *reinterpret_cast<const v8h*>(&sA[rowA1 + ko]);
        v8h a1hi = *reinterpret_cast<const v8h*>(&sA[rowA1 + ko + 8]);
        v16h a0 = __builtin_shufflevector(a0lo, a0hi,
                    0,1,2,3,4,5,6,7,8,9,10,11,12,13,14,15);
        v16h a1 = __builtin_shufflevector(a1lo, a1hi,
                    0,1,2,3,4,5,6,7,8,9,10,11,12,13,14,15);

        #pragma unroll
        for (int nt = 0; nt < NTILES; ++nt) {
            const _Float16* bp =
                bfrag + ((size_t)(nt * KSTEPS + kk) * 32 + lane) * 16;
            v8h blo = *reinterpret_cast<const v8h*>(bp);
            v8h bhi = *reinterpret_cast<const v8h*>(bp + 8);
            v16h bm = __builtin_shufflevector(blo, bhi,
                        0,1,2,3,4,5,6,7,8,9,10,11,12,13,14,15);
            acc[0][nt] = __builtin_amdgcn_wmma_f32_16x16x32_f16(
                false, a0, false, bm, (short)0, acc[0][nt], false, false);
            acc[1][nt] = __builtin_amdgcn_wmma_f32_16x16x32_f16(
                false, a1, false, bm, (short)0, acc[1][nt], false, false);
        }
    }

    // ---- Phase 3: epilogue -----------------------------------------------
    // C layout: lanes 0-15 -> N=lane, M=r ; lanes 16-31 -> N=lane-16, M=8+r
    #pragma unroll
    for (int mt = 0; mt < 2; ++mt) {
        const int mbase = (wave * 2 + mt) * 16;
        #pragma unroll
        for (int r = 0; r < 8; ++r) {
            const int prow = mbase + r + (hi ? 8 : 0);   // pixel within block
            const float inv = sInv[prow];
            float m4 = 2.0f - 2.0f * acc[mt][0][r] * inv;
            #pragma unroll
            for (int nt = 1; nt < NTILES; ++nt) {
                float d2 = 2.0f - 2.0f * acc[mt][nt][r] * inv;
                m4 = fminf(m4, d2);
            }
            sRed[prow * 16 + l15] = m4;   // min over this lane's 4 proxy tiles
        }
    }
    __syncthreads();

    // 16 partial mins per pixel -> final min, sqrt
    {
        float m = sRed[t * 16];
        #pragma unroll
        for (int j = 1; j < 16; ++j)
            m = fminf(m, sRed[t * 16 + j]);
        out[(long)blk * MPIX + t] = sqrtf(fmaxf(m, 0.0f));
    }
}

// ---------------------------------------------------------------------------
extern "C" void kernel_launch(void* const* d_in, const int* in_sizes, int n_in,
                              void* d_out, int out_size, void* d_ws, size_t ws_size,
                              hipStream_t stream) {
    const float* x    = (const float*)d_in[0];     // (32, 384, 128, 128) f32
    const float* prox = (const float*)d_in[1];     // (64, 384) f32
    float* out        = (float*)d_out;             // (32, 128, 128) f32
    _Float16* bfrag   = (_Float16*)d_ws;           // 48 KB of B fragments

    proxy_prep_kernel<<<1, 64, 0, stream>>>(prox, bfrag);

    const int n_pixels = in_sizes[0] / D_DIM;      // 524288
    const int n_blocks = n_pixels / MPIX;          // 4096
    proxy_min_dist_kernel<<<n_blocks, MPIX, 0, stream>>>(x, bfrag, out);
}
